// Attention_61332132987532
// MI455X (gfx1250) — matrix operand order repaired and checked by reading, not compiled, not run on hardware
//
#include <hip/hip_runtime.h>
#include <math.h>

// Problem constants (match reference)
#define BB   64
#define SS   4096
#define AA   256
#define DDIM 1024
#define CTXD 1024
#define ST   64          // S-tile per workgroup
#define NTHREADS 512     // 16 waves (wave32)
#define KSTEPS (CTXD / 32)

typedef __attribute__((ext_vector_type(16))) __bf16         v16bf;
typedef __attribute__((ext_vector_type(8)))  float          v8f;
typedef __attribute__((ext_vector_type(16))) unsigned short v16u;
typedef __attribute__((ext_vector_type(8)))  unsigned short us8;
typedef __attribute__((ext_vector_type(4)))  unsigned int   v4u;
typedef __attribute__((ext_vector_type(8)))  int            v8i;
typedef __attribute__((ext_vector_type(4)))  int            v4i;

union FragU {
    v16u u;
    us8  h[2];
};

__device__ __forceinline__ unsigned short f2bf(float f) {
    unsigned int u = __float_as_uint(f);
    unsigned int r = u + 0x7FFFu + ((u >> 16) & 1u);   // round-to-nearest-even
    return (unsigned short)(r >> 16);
}

// ---------------------------------------------------------------------------
// TDM: load a [256 x 32] bf16 tile (row stride 1024 elements) from global
// (pre-converted U) into LDS. 2D descriptor, data_size=2B, type=2 ("image").
// 6-arg builtin variant (clang-23 toolchain).
// ---------------------------------------------------------------------------
__device__ __forceinline__ void tdm_load_U(const unsigned short* Ubf, int kk,
                                           unsigned ldsByteAddr) {
    unsigned long long ga = (unsigned long long)(const void*)Ubf +
                            (unsigned long long)kk * 2ull;   // tile start (bytes)
    // Group0: [1:0]=count=1 | lds_addr | global_addr[56:0] | type=2 @ [127:126]
    v4u g0 = { 1u,
               ldsByteAddr,
               (unsigned)ga,
               (unsigned)((ga >> 32) & 0x01FFFFFFull) | 0x80000000u };
    // Group1: data_size=1(2B)<<16 | tensor_dim0=1024 | tensor_dim1=256 |
    //         tile_dim0=32 | tile_dim1=256 | tensor_dim0_stride=1024
    v8i g1 = { (int)(1u << 16),          // wg_mask=0, data_size=2B
               (int)(1024u << 16),       // tensor_dim0[15:0] in [31:16]
               (int)(256u << 16),        // tensor_dim0[31:16]=0, tensor_dim1[15:0]
               (int)(32u << 16),         // tensor_dim1[31:16]=0, tile_dim0=32
               (int)256,                 // tile_dim1=256, tile_dim2=0
               (int)1024,                // tensor_dim0_stride[31:0]
               0, 0 };                   // stride hi / tensor_dim1_stride = 0
    v4i gz4 = { 0, 0, 0, 0 };            // groups 2/3 unused (2D tile)
    v8i gz8 = { 0, 0, 0, 0, 0, 0, 0, 0 };
    __builtin_amdgcn_tensor_load_to_lds(g0, g1, gz4, gz4, gz8, 0);
}

// ---------------------------------------------------------------------------
// Kernel 0: zero the accumulation workspace (gnum [B*CTX] + gden [B])
// ---------------------------------------------------------------------------
__global__ void attn_zero_kernel(float* p, int n) {
    int i = blockIdx.x * blockDim.x + threadIdx.x;
    if (i < n) p[i] = 0.0f;
}

// ---------------------------------------------------------------------------
// Kernel 1a: convert U [A,CTX] f32 -> bf16 once (removes per-WG conversion)
// ---------------------------------------------------------------------------
__global__ void attn_cvtU_kernel(const float* __restrict__ U,
                                 unsigned short* __restrict__ Ubf) {
    int i = blockIdx.x * blockDim.x + threadIdx.x;   // 65536 threads x 4 elems
    float4 u4 = *(const float4*)&U[(size_t)i * 4];
    ushort4 b;
    b.x = f2bf(u4.x); b.y = f2bf(u4.y); b.z = f2bf(u4.z); b.w = f2bf(u4.w);
    *(ushort4*)&Ubf[(size_t)i * 4] = b;
}

// ---------------------------------------------------------------------------
// Kernel 1b: dec_proj[b][a] = prev[b] . W[a] + b_W[a]   (tiny, 34 MFLOP)
// ---------------------------------------------------------------------------
__global__ void attn_decproj_kernel(const float* __restrict__ prev,
                                    const float* __restrict__ W,
                                    const float* __restrict__ bW,
                                    float* __restrict__ dp) {
    __shared__ __align__(16) float prevSh[DDIM];
    const int b = blockIdx.x;
    const int t = threadIdx.x;           // 256 threads = one output 'a' each
    *(float4*)&prevSh[t * 4] = *(const float4*)&prev[(size_t)b * DDIM + t * 4];
    __syncthreads();
    float acc = bW[t];
    const float4* wr = (const float4*)&W[(size_t)t * DDIM];
#pragma unroll 4
    for (int k4 = 0; k4 < DDIM / 4; ++k4) {
        float4 w = wr[k4];
        acc += w.x * prevSh[k4 * 4 + 0] + w.y * prevSh[k4 * 4 + 1] +
               w.z * prevSh[k4 * 4 + 2] + w.w * prevSh[k4 * 4 + 3];
    }
    dp[b * AA + t] = acc;
}

// ---------------------------------------------------------------------------
// Kernel 2: fused  U-projection GEMM (bf16 WMMA, TDM-staged B operand)
//           -> tanh -> v-dot -> exp -> partial weighted context sum.
// grid = (S/ST, B), block = 512 threads (16 waves).
// ---------------------------------------------------------------------------
__global__ void __launch_bounds__(NTHREADS)
attn_main_kernel(const float* __restrict__ enc,            // [B,S,CTX] f32
                 const unsigned short* __restrict__ Ubf,   // [A,CTX] bf16
                 const float* __restrict__ bU,             // [A]
                 const float* __restrict__ v,              // [A]
                 const float* __restrict__ dp,             // [B,A]
                 float* __restrict__ gnum,                 // [B,CTX] accum
                 float* __restrict__ gden)                 // [B]     accum
{
    __shared__ __align__(16) unsigned short encB[ST * 32];     // bf16 enc slice
    __shared__ __align__(16) unsigned short Ub[2][AA * 32];    // TDM dest (dbl buf)
    __shared__ float dpbu[AA];
    __shared__ float vsh[AA];
    __shared__ float energy[ST];
    __shared__ float exsh[ST];
    __shared__ float denSh;

    const int b   = blockIdx.y;
    const int s0  = blockIdx.x * ST;
    const int t   = threadIdx.x;
    const int lane   = t & 31;
    const int waveId = t >> 5;
    const int m0     = (waveId & 3) * 16;   // wave's M (s-row) tile
    const int n0base = (waveId >> 2) * 64;  // wave's N (a-col) strip (4 tiles)

    // init per-block state
    if (t < AA) {
        dpbu[t] = dp[b * AA + t] + bU[t];
        vsh[t]  = v[t];
    }
    if (t < ST) energy[t] = 0.0f;
    if (t == 0) denSh = 0.0f;

    v8f acc0 = {}, acc1 = {}, acc2 = {}, acc3 = {};

    const size_t encRowBase = ((size_t)b * SS + s0) * CTXD;

    // flat-LDS byte addresses of the two U buffers (low 32 bits of generic VA)
    const unsigned ldsU0 = (unsigned)(unsigned long long)(void*)&Ub[0][0];
    const unsigned ldsU1 = (unsigned)(unsigned long long)(void*)&Ub[1][0];

    // staging indices: each thread stages one float4 of the enc slice
    const int sr  = t >> 3;           // 0..63  enc row
    const int sc4 = (t & 7) << 2;     // 0,4,..28 col within 32-slice

    // prime the TDM pipeline with the first U slice
    if (t == 0) tdm_load_U(Ubf, 0, ldsU0);

    for (int it = 0; it < KSTEPS; ++it) {
        const int kk = it * 32;

        // ---- stage enc[64][32] slice to LDS as bf16 (RNE truncation) -----
        {
            float4 e4 = *(const float4*)&enc[encRowBase + (size_t)sr * CTXD + kk + sc4];
            ushort4 eb;
            eb.x = f2bf(e4.x); eb.y = f2bf(e4.y); eb.z = f2bf(e4.z); eb.w = f2bf(e4.w);
            *(ushort4*)&encB[sr * 32 + sc4] = eb;
            if (kk + 32 < CTXD)
                __builtin_prefetch(&enc[encRowBase + (size_t)sr * CTXD + kk + 32 + sc4], 0, 3);
        }

        // ---- pipeline TDM: issue next slice, wait for current ------------
        if (t == 0) {
            if (it + 1 < KSTEPS) {
                tdm_load_U(Ubf, kk + 32, (it & 1) ? ldsU0 : ldsU1);  // next buf
                __builtin_amdgcn_s_wait_tensorcnt(1);                 // cur done
            } else {
                __builtin_amdgcn_s_wait_tensorcnt(0);
            }
        }
        __syncthreads();

        const unsigned short* Ucur = Ub[it & 1];

        // ---- A fragment (16x32 bf16): lane<16 -> K {0..7,16..23},
        //      lane>=16 -> +8 ; row = m0 + (lane&15) -----------------------
        const int rowA = m0 + (lane & 15);
        const int kbA  = (lane >= 16) ? 8 : 0;
        FragU fa;
        fa.h[0] = *(const us8*)&encB[rowA * 32 + kbA];        // K h0..7
        fa.h[1] = *(const us8*)&encB[rowA * 32 + kbA + 16];   // K h8..15
        v16bf aw = __builtin_bit_cast(v16bf, fa.u);

        // ---- B fragments (32x16 bf16): col = n + (lane&15),
        //      lane<16 -> K {0..7,16..23}, lane>=16 -> +8 -------------------
        const int kbB = (lane >= 16) ? 8 : 0;
#pragma unroll
        for (int nt = 0; nt < 4; ++nt) {
            const int ncol = n0base + nt * 16 + (lane & 15);
            FragU fb;
            fb.h[0] = *(const us8*)&Ucur[ncol * 32 + kbB];
            fb.h[1] = *(const us8*)&Ucur[ncol * 32 + kbB + 16];
            v16bf bw = __builtin_bit_cast(v16bf, fb.u);
            if      (nt == 0) acc0 = __builtin_amdgcn_wmma_f32_16x16x32_bf16(false, aw, false, bw, (short)0, acc0, false, false);
            else if (nt == 1) acc1 = __builtin_amdgcn_wmma_f32_16x16x32_bf16(false, aw, false, bw, (short)0, acc1, false, false);
            else if (nt == 2) acc2 = __builtin_amdgcn_wmma_f32_16x16x32_bf16(false, aw, false, bw, (short)0, acc2, false, false);
            else              acc3 = __builtin_amdgcn_wmma_f32_16x16x32_bf16(false, aw, false, bw, (short)0, acc3, false, false);
        }
        __syncthreads();
    }

    // ---- epilogue: energy[m] += sum_n v[n]*tanh(dpbu[n] + proj[m][n]) -----
    // C layout: VGPR r, lanes 0-15 -> M=r, lanes 16-31 -> M=r+8; N = lane&15
    const int nlo = lane & 15;
    const int mhi = (lane >= 16) ? 8 : 0;
#pragma unroll
    for (int r = 0; r < 8; ++r) {
        const int m = m0 + r + mhi;
        float partial =
            vsh[n0base +  0 + nlo] * tanhf(dpbu[n0base +  0 + nlo] + acc0[r]) +
            vsh[n0base + 16 + nlo] * tanhf(dpbu[n0base + 16 + nlo] + acc1[r]) +
            vsh[n0base + 32 + nlo] * tanhf(dpbu[n0base + 32 + nlo] + acc2[r]) +
            vsh[n0base + 48 + nlo] * tanhf(dpbu[n0base + 48 + nlo] + acc3[r]);
        atomicAdd(&energy[m], partial);               // ds_add_f32
    }
    __syncthreads();

    if (t < ST) {
        float e = expf(energy[t]);                    // faithful: no max-sub
        exsh[t] = e;
        atomicAdd(&denSh, e);
    }
    __syncthreads();

    // ---- partial context: num[c] += sum_m exp(e_m) * enc[b][s0+m][c] ------
    // tile was just streamed -> hits 192MB L2
    for (int cc = t; cc < CTXD; cc += NTHREADS) {
        float num = 0.0f;
#pragma unroll 8
        for (int m = 0; m < ST; ++m)
            num += exsh[m] * enc[encRowBase + (size_t)m * CTXD + cc];
        unsafeAtomicAdd(&gnum[b * CTXD + cc], num);   // global_atomic_add_f32
    }
    if (t == 0) unsafeAtomicAdd(&gden[b], denSh);
}

// ---------------------------------------------------------------------------
// Kernel 3: c[b][c] = gnum[b][c] / gden[b]
// ---------------------------------------------------------------------------
__global__ void attn_finalize_kernel(const float* __restrict__ gnum,
                                     const float* __restrict__ gden,
                                     float* __restrict__ out) {
    int i = blockIdx.x * blockDim.x + threadIdx.x;   // 65536
    out[i] = gnum[i] / gden[i >> 10];
}

// ---------------------------------------------------------------------------
extern "C" void kernel_launch(void* const* d_in, const int* in_sizes, int n_in,
                              void* d_out, int out_size, void* d_ws, size_t ws_size,
                              hipStream_t stream) {
    const float* prev = (const float*)d_in[0];  // [B,DD]
    const float* enc  = (const float*)d_in[1];  // [B,S,CTX]
    const float* W    = (const float*)d_in[2];  // [A,DD]
    const float* bW   = (const float*)d_in[3];  // [A]
    const float* U    = (const float*)d_in[4];  // [A,CTX]
    const float* bU   = (const float*)d_in[5];  // [A]
    const float* v    = (const float*)d_in[6];  // [A]
    float* out = (float*)d_out;

    float* wsf  = (float*)d_ws;
    float* gnum = wsf;                                   // 65536 f32
    float* gden = wsf + BB * CTXD;                       // 64 f32
    float* dp   = wsf + BB * CTXD + BB;                  // 16384 f32
    unsigned short* Ubf =
        (unsigned short*)(wsf + BB * CTXD + BB + BB * AA);  // 262144 bf16

    const int nzero = BB * CTXD + BB;
    attn_zero_kernel<<<(nzero + 255) / 256, 256, 0, stream>>>(wsf, nzero);
    attn_cvtU_kernel<<<(AA * CTXD / 4 + 255) / 256, 256, 0, stream>>>(U, Ubf);
    attn_decproj_kernel<<<BB, AA, 0, stream>>>(prev, W, bW, dp);
    attn_main_kernel<<<dim3(SS / ST, BB), NTHREADS, 0, stream>>>(enc, Ubf, bU, v, dp, gnum, gden);
    attn_finalize_kernel<<<(BB * CTXD) / 256, 256, 0, stream>>>(gnum, gden, out);
}
